// CifarModel_11398843203602
// MI455X (gfx1250) — matrix operand order repaired and checked by reading, not compiled, not run on hardware
//
#include <hip/hip_runtime.h>
#include <hip/hip_fp16.h>

typedef _Float16 v16h __attribute__((ext_vector_type(16)));
typedef _Float16 v8h  __attribute__((ext_vector_type(8)));
typedef float    v8f  __attribute__((ext_vector_type(8)));
typedef float    v4f  __attribute__((ext_vector_type(4)));

union HV16 { v16h v; v8h h[2]; _Float16 e[16]; };

// ---------------------------------------------------------------------------
// act0 init: zero the 8-channel padded NHWC f16 buffer; if stage != 2, copy
// x (NHWC f32, 3ch) straight in (reference bypasses the block-linear then).
// ---------------------------------------------------------------------------
__global__ void k_prep_zero(const float* __restrict__ x, const int* __restrict__ stage,
                            _Float16* __restrict__ act0, int npix) {
    int p = blockIdx.x * blockDim.x + threadIdx.x;
    if (p >= npix) return;
    v8h z = {};
    if (*stage != 2) {
        const float* s = x + (size_t)p * 3;
        z[0] = (_Float16)s[0]; z[1] = (_Float16)s[1]; z[2] = (_Float16)s[2];
    }
    *(v8h*)(act0 + (size_t)p * 8) = z;
}

// W_lin (48x48 f32) -> f16 padded to 48x64 (K pad = 0)
__global__ void k_repack_wlin(const float* __restrict__ w, _Float16* __restrict__ out) {
    int i = blockIdx.x * blockDim.x + threadIdx.x;
    if (i >= 48 * 64) return;
    int o = i >> 6, k = i & 63;
    out[i] = (k < 48) ? (_Float16)w[o * 48 + k] : (_Float16)0.0f;
}

// conv1 OIHW(32,3,3,3) f32 -> row-packed f16: out[ky][co][32], K = kx*8 + c
// (three horizontal taps fused into one K=24 (pad 32) WMMA per kernel row)
__global__ void k_repack_conv1p(const float* __restrict__ w, _Float16* __restrict__ out) {
    int i = blockIdx.x * blockDim.x + threadIdx.x;
    if (i >= 3 * 32 * 32) return;
    int ky = i / (32 * 32); int r = i % (32 * 32);
    int co = r >> 5; int k = r & 31;
    int kx = k >> 3, c = k & 7;
    float v = 0.0f;
    if (k < 24 && c < 3) v = w[((co * 3 + c) * 3 + ky) * 3 + kx];
    out[i] = (_Float16)v;
}

// conv OIHW f32 -> tap-major f16: wt[tap][co][ci_padded]
__global__ void k_repack_conv(const float* __restrict__ w, _Float16* __restrict__ out,
                              int CO, int CIN, int KP) {
    int i = blockIdx.x * blockDim.x + threadIdx.x;
    int n = 9 * CO * KP;
    if (i >= n) return;
    int tap = i / (CO * KP); int r = i % (CO * KP);
    int co = r / KP; int ci = r % KP;
    float v = 0.0f;
    if (ci < CIN) { int ky = tap / 3, kx = tap % 3; v = w[((co * CIN + ci) * 3 + ky) * 3 + kx]; }
    out[i] = (_Float16)v;
}

// ---------------------------------------------------------------------------
// Stage-2 block linear: for every 4x4x3 spatial block, out48 = W_lin @ in48.
// (The fixed permutation reduces to "apply in place" in spatial coordinates.)
// D = A(16x32 = W rows) x B(32x16 = 16 blocks), two WMMAs for K=48 (pad 64).
// B operand assembled from 16B-aligned float4 runs of the 4x4x3 block rows.
// ---------------------------------------------------------------------------
__device__ __forceinline__ void cvt4(_Float16* dst, const float* src) {
    v4f v = *(const v4f*)src;
    dst[0] = (_Float16)v[0]; dst[1] = (_Float16)v[1];
    dst[2] = (_Float16)v[2]; dst[3] = (_Float16)v[3];
}

__global__ __launch_bounds__(256) void k_blocklinear(
        const float* __restrict__ x, const _Float16* __restrict__ wl,
        const int* __restrict__ stage, _Float16* __restrict__ act0, int nb16) {
    if (*stage != 2) return;
    int wid  = (blockIdx.x * blockDim.x + threadIdx.x) >> 5;
    int lane = threadIdx.x & 31;
    int total = nb16 * 3;
    if (wid >= total) return;
    int g = wid / 3, tile = wid % 3;
    int col = lane & 15;
    int hi  = lane >> 4;
    int blk = g * 16 + col;
    int bat = blk >> 6, nb = blk & 63;
    int ib = nb >> 3, jb = nb & 7;
    const float* xb = x + ((((size_t)bat * 32 + ib * 4) * 32) + jb * 4) * 3;  // row stride 96 f32

    HV16 b0 = {}, b1 = {};
    if (hi == 0) {
        // K 0..15  = block elems 0..15 : row0[0..11], row1[0..3]
        cvt4(&b0.e[0],  xb + 0);  cvt4(&b0.e[4],  xb + 4);  cvt4(&b0.e[8], xb + 8);
        cvt4(&b0.e[12], xb + 96);
        // K 32..47 = block elems 32..47: row2[8..11], row3[0..11]
        cvt4(&b1.e[0],  xb + 2 * 96 + 8);
        cvt4(&b1.e[4],  xb + 3 * 96 + 0); cvt4(&b1.e[8], xb + 3 * 96 + 4);
        cvt4(&b1.e[12], xb + 3 * 96 + 8);
    } else {
        // K 16..31 = block elems 16..31: row1[4..11], row2[0..7]
        cvt4(&b0.e[0],  xb + 96 + 4);  cvt4(&b0.e[4],  xb + 96 + 8);
        cvt4(&b0.e[8],  xb + 2 * 96 + 0); cvt4(&b0.e[12], xb + 2 * 96 + 4);
        // K 48..63 padding -> zero
    }

    const _Float16* wr = wl + (size_t)(tile * 16 + col) * 64;
    int akb = hi * 16;
    HV16 a0, a1;
    a0.h[0] = *(const v8h*)(wr + akb);
    a0.h[1] = *(const v8h*)(wr + akb + 8);
    a1.h[0] = *(const v8h*)(wr + 32 + akb);
    a1.h[1] = *(const v8h*)(wr + 32 + akb + 8);

    v8f acc = {};
    acc = __builtin_amdgcn_wmma_f32_16x16x32_f16(false, a0.v, false, b0.v, (short)0, acc, false, false);
    acc = __builtin_amdgcn_wmma_f32_16x16x32_f16(false, a1.v, false, b1.v, (short)0, acc, false, false);

    // D: lane holds N = col (this block), M = tile*16 + 8*hi + r
#pragma unroll
    for (int r = 0; r < 8; ++r) {
        int o = tile * 16 + hi * 8 + r;          // output feature = ir*12 + jc*3 + c
        int ir = o / 12, t = o % 12, jc = t / 3, c = t % 3;
        size_t idx = ((((size_t)bat * 32 + ib * 4 + ir) * 32) + jb * 4 + jc) * 8 + c;
        act0[idx] = (_Float16)acc[r];
    }
}

// ---------------------------------------------------------------------------
// Shared epilogue: bias + ReLU -> per-wave LDS tiles -> in-wave 2x2 maxpool,
// wide contiguous channel stores (MT*2 halves per lane).
// ---------------------------------------------------------------------------
template <int MT, int CO, int H, int W>
__device__ __forceinline__ void bias_relu_pool_store(
        const v8f* acc, const float* __restrict__ bias, _Float16* __restrict__ out,
        _Float16 (*tile)[16][16], int n, int y0, int x0, int mt0, int lane) {
    int col = lane & 15, hi = lane >> 4, mb = hi * 8;
#pragma unroll
    for (int m = 0; m < MT; ++m) {
        v8h hv;
#pragma unroll
        for (int r = 0; r < 8; ++r) {
            float v = acc[m][r] + bias[(mt0 * MT + m) * 16 + mb + r];
            hv[r] = (_Float16)(v > 0.0f ? v : 0.0f);
        }
        *(v8h*)&tile[m][col][mb] = hv;
    }
    __asm__ volatile("" ::: "memory");   // keep DS store->load order; DS in-order per wave

    constexpr int CPL = MT * 2;          // channels per lane (within one 16-ch tile)
    int w = lane & 3, chbase = (lane >> 2) * CPL;
    int m = chbase >> 4, inner = chbase & 15;
    float tmp[CPL];
#pragma unroll
    for (int c = 0; c < CPL; ++c) tmp[c] = 0.0f;   // post-ReLU values are >= 0
#pragma unroll
    for (int t = 0; t < 4; ++t) {
        int Np = ((t >> 1) << 3) | (w << 1) | (t & 1);
        const _Float16* p = &tile[m][Np][inner];
#pragma unroll
        for (int c = 0; c < CPL; ++c) tmp[c] = fmaxf(tmp[c], (float)p[c]);
    }
    int X = (x0 >> 1) + w, Y = y0 >> 1;
    _Float16* op = out + ((((size_t)n * (H / 2) + Y) * (W / 2)) + X) * CO + mt0 * MT * 16 + chbase;
#pragma unroll
    for (int c = 0; c < CPL; ++c) op[c] = (_Float16)tmp[c];
}

// ---------------------------------------------------------------------------
// conv1 (3->32ch, 32x32) fused conv+bias+ReLU+pool, row-packed K:
// one WMMA per kernel row (K = 3 taps x 8ch = 24, pad 32), MT=2 covers all co.
// ---------------------------------------------------------------------------
__global__ __launch_bounds__(256) void k_conv1_pool(
        const _Float16* __restrict__ in, const _Float16* __restrict__ wt,
        const float* __restrict__ bias, _Float16* __restrict__ out, int nimg) {
    constexpr int H = 32, W = 32, CO = 32, MT = 2, G = (W / 8) * (H / 2);
    __shared__ _Float16 lds[8][MT][16][16];
    int wid  = (blockIdx.x * blockDim.x + threadIdx.x) >> 5;
    int lane = threadIdx.x & 31;
    int wv   = threadIdx.x >> 5;
    if (wid >= nimg * G) return;
    int n = wid / G, gidx = wid % G;
    int y0 = (gidx / (W / 8)) * 2, x0 = (gidx % (W / 8)) * 8;
    int col = lane & 15, hi = lane >> 4;
    int py = y0 + (col >> 3), px = x0 + (col & 7);
    int akb = hi * 16;

    __builtin_prefetch(in + (((size_t)n * H + py) * W + px) * 8, 0, 0);

    v8f acc[MT] = {};
#pragma unroll
    for (int ky = 0; ky < 3; ++ky) {
        int ys = py + ky - 1;
        bool rv = (ys >= 0) & (ys < H);
        const _Float16* rowp = in + (((size_t)n * H + ys) * W) * 8;
        HV16 bb = {};
        if (rv) {
            if (hi == 0) {                        // K0..7 = dx-1, K8..15 = dx0
                if (px - 1 >= 0) bb.h[0] = *(const v8h*)(rowp + (px - 1) * 8);
                bb.h[1] = *(const v8h*)(rowp + px * 8);
            } else {                              // K16..23 = dx+1, K24..31 pad
                if (px + 1 < W) bb.h[0] = *(const v8h*)(rowp + (px + 1) * 8);
            }
        }
#pragma unroll
        for (int m = 0; m < MT; ++m) {
            const _Float16* wq = wt + (size_t)(ky * CO + m * 16 + col) * 32 + akb;
            HV16 a;
            a.h[0] = *(const v8h*)(wq);
            a.h[1] = *(const v8h*)(wq + 8);
            acc[m] = __builtin_amdgcn_wmma_f32_16x16x32_f16(false, a.v, false, bb.v,
                                                            (short)0, acc[m], false, false);
        }
    }
    bias_relu_pool_store<MT, CO, H, W>(acc, bias, out, lds[wv], n, y0, x0, 0, lane);
}

// ---------------------------------------------------------------------------
// Generic fused conv3x3(SAME)+bias+ReLU+pool, NHWC f16, MT co-tiles per wave:
// B activation fragment loaded once per (tap,chunk) and fed to MT WMMAs.
// ---------------------------------------------------------------------------
template <int CO, int CSTORE, int KCH, int H, int W, int MT>
__global__ __launch_bounds__(256) void k_conv_pool(
        const _Float16* __restrict__ in, const _Float16* __restrict__ wt,
        const float* __restrict__ bias, _Float16* __restrict__ out, int nimg) {
    constexpr int GX = W / 8, GY = H / 2, G = GX * GY, WGRP = (CO / 16) / MT;
    __shared__ _Float16 lds[8][MT][16][16];
    int wid  = (blockIdx.x * blockDim.x + threadIdx.x) >> 5;
    int lane = threadIdx.x & 31;
    int wv   = threadIdx.x >> 5;
    if (wid >= nimg * G * WGRP) return;
    int n = wid / (G * WGRP); int r0 = wid % (G * WGRP);
    int gidx = r0 / WGRP; int mt0 = r0 % WGRP;
    int y0 = (gidx / GX) * 2, x0 = (gidx % GX) * 8;
    int col = lane & 15, hi = lane >> 4;
    int py = y0 + (col >> 3), px = x0 + (col & 7);
    int akb = hi * 16;

    __builtin_prefetch(in + (((size_t)n * H + py) * W + px) * CSTORE, 0, 0);

    v8f acc[MT] = {};
#pragma unroll
    for (int tap = 0; tap < 9; ++tap) {
        int dy = tap / 3 - 1, dx = tap % 3 - 1;
        int ys = py + dy, xs = px + dx;
        bool inb = (ys >= 0) & (ys < H) & (xs >= 0) & (xs < W);
        const _Float16* src = in + (((size_t)n * H + ys) * W + xs) * CSTORE;
#pragma unroll
        for (int q = 0; q < KCH; ++q) {
            HV16 bb = {};
            if (inb) {
                int cb = q * 32 + hi * 16;
                bb.h[0] = *(const v8h*)(src + cb);
                bb.h[1] = *(const v8h*)(src + cb + 8);
            }
#pragma unroll
            for (int m = 0; m < MT; ++m) {
                const _Float16* wq = wt +
                    (size_t)(tap * CO + (mt0 * MT + m) * 16 + col) * (KCH * 32) + q * 32 + akb;
                HV16 a;
                a.h[0] = *(const v8h*)(wq);
                a.h[1] = *(const v8h*)(wq + 8);
                acc[m] = __builtin_amdgcn_wmma_f32_16x16x32_f16(false, a.v, false, bb.v,
                                                                (short)0, acc[m], false, false);
            }
        }
    }
    bias_relu_pool_store<MT, CO, H, W>(acc, bias, out, lds[wv], n, y0, x0, mt0, lane);
}

// ---------------------------------------------------------------------------
// FC (2048 -> 10) + log_softmax. One wave per image. act3 is NHWC(4,4,128) f16;
// reference flattens NCHW so weight index j = c*16 + y*4 + x.
// ---------------------------------------------------------------------------
__global__ __launch_bounds__(256) void k_fc_logsoftmax(
        const _Float16* __restrict__ act3, const float* __restrict__ fw,
        const float* __restrict__ fb, float* __restrict__ out, int nimg) {
    int wid  = (blockIdx.x * blockDim.x + threadIdx.x) >> 5;
    int lane = threadIdx.x & 31;
    if (wid >= nimg) return;
    const _Float16* h = act3 + (size_t)wid * 2048;
    float acc[10];
#pragma unroll
    for (int o = 0; o < 10; ++o) acc[o] = 0.0f;
    for (int j = lane; j < 2048; j += 32) {
        int c = j >> 4, s = j & 15;              // s = y*4 + x
        float hv = (float)h[s * 128 + c];
#pragma unroll
        for (int o = 0; o < 10; ++o) acc[o] += hv * fw[o * 2048 + j];
    }
#pragma unroll
    for (int o = 0; o < 10; ++o)
#pragma unroll
        for (int off = 16; off > 0; off >>= 1)
            acc[o] += __shfl_down(acc[o], off, 32);
    if (lane == 0) {
        float l[10], m = -3.0e38f;
#pragma unroll
        for (int o = 0; o < 10; ++o) { l[o] = acc[o] + fb[o]; m = fmaxf(m, l[o]); }
        float s = 0.0f;
#pragma unroll
        for (int o = 0; o < 10; ++o) s += __expf(l[o] - m);
        float lse = __logf(s);
#pragma unroll
        for (int o = 0; o < 10; ++o) out[(size_t)wid * 10 + o] = l[o] - m - lse;
    }
}

// ---------------------------------------------------------------------------
extern "C" void kernel_launch(void* const* d_in, const int* in_sizes, int n_in,
                              void* d_out, int out_size, void* d_ws, size_t ws_size,
                              hipStream_t stream) {
    const float* x     = (const float*)d_in[0];
    const float* W_lin = (const float*)d_in[1];
    const float* c1w   = (const float*)d_in[2];
    const float* c1b   = (const float*)d_in[3];
    const float* c2w   = (const float*)d_in[4];
    const float* c2b   = (const float*)d_in[5];
    const float* c3w   = (const float*)d_in[6];
    const float* c3b   = (const float*)d_in[7];
    const float* fcw   = (const float*)d_in[8];
    const float* fcb   = (const float*)d_in[9];
    const int*   stage = (const int*)d_in[10];
    float* out = (float*)d_out;
    int B = in_sizes[0] / 3072;   // (B,32,32,3)

    char* ws = (char*)d_ws;
    size_t off = 0;
    auto alloc = [&](size_t bytes) -> char* {
        char* p = ws + off;
        off = (off + bytes + 255) & ~(size_t)255;
        return p;
    };
    _Float16* wl16 = (_Float16*)alloc((size_t)48 * 64 * 2);
    _Float16* wt1  = (_Float16*)alloc((size_t)3 * 32 * 32 * 2);
    _Float16* wt2  = (_Float16*)alloc((size_t)9 * 64 * 32 * 2);
    _Float16* wt3  = (_Float16*)alloc((size_t)9 * 128 * 64 * 2);
    _Float16* act0 = (_Float16*)alloc((size_t)B * 32 * 32 * 8 * 2);
    _Float16* act1 = (_Float16*)alloc((size_t)B * 16 * 16 * 32 * 2);
    _Float16* act2 = (_Float16*)alloc((size_t)B * 8 * 8 * 64 * 2);
    _Float16* act3 = (_Float16*)alloc((size_t)B * 4 * 4 * 128 * 2);
    (void)ws_size; (void)n_in; (void)out_size;

    k_repack_wlin<<<(48 * 64 + 255) / 256, 256, 0, stream>>>(W_lin, wl16);
    k_repack_conv1p<<<(3 * 32 * 32 + 255) / 256, 256, 0, stream>>>(c1w, wt1);
    k_repack_conv<<<(9 * 64 * 32 + 255) / 256, 256, 0, stream>>>(c2w, wt2, 64, 32, 32);
    k_repack_conv<<<(9 * 128 * 64 + 255) / 256, 256, 0, stream>>>(c3w, wt3, 128, 64, 64);

    int npix = B * 32 * 32;
    k_prep_zero<<<(npix + 255) / 256, 256, 0, stream>>>(x, stage, act0, npix);

    int nb16 = B * 64 / 16;
    {
        long long waves = (long long)nb16 * 3;
        k_blocklinear<<<(int)((waves * 32 + 255) / 256), 256, 0, stream>>>(x, wl16, stage, act0, nb16);
    }
    {
        long long waves = (long long)B * 64;                 // G=64, all 32 co per wave
        k_conv1_pool<<<(int)((waves * 32 + 255) / 256), 256, 0, stream>>>(act0, wt1, c1b, act1, B);
    }
    {
        long long waves = (long long)B * 16;                 // G=16, WGRP=1 (MT=4)
        k_conv_pool<64, 32, 1, 16, 16, 4>
            <<<(int)((waves * 32 + 255) / 256), 256, 0, stream>>>(act1, wt2, c2b, act2, B);
    }
    {
        long long waves = (long long)B * 4 * 2;              // G=4, WGRP=2 (MT=4)
        k_conv_pool<128, 64, 2, 8, 8, 4>
            <<<(int)((waves * 32 + 255) / 256), 256, 0, stream>>>(act2, wt3, c3b, act3, B);
    }
    k_fc_logsoftmax<<<(B * 32 + 255) / 256, 256, 0, stream>>>(act3, fcw, fcb, out, B);
}
